// ChaoticAttentionLayer_71683004170564
// MI455X (gfx1250) — compile-verified, matrix operand order checked by reading, not run on hardware
//
#include <hip/hip_runtime.h>

// ---------------------------------------------------------------------------
// ChaoticAttentionLayer for MI455X (gfx1250, wave32, WMMA bf16, async-LDS)
// B=4, T=S=2048, E=512, H=8, D=64
// ---------------------------------------------------------------------------

typedef __attribute__((ext_vector_type(16))) __bf16 v16bf;
typedef __attribute__((ext_vector_type(8)))  __bf16 v8bf;
typedef __attribute__((ext_vector_type(8)))  float  v8f;
typedef __attribute__((ext_vector_type(4)))  float  v4f;

#define E_  512
#define T_  2048
#define H_  8
#define D_  64

union V16 { v16bf v; v8bf h[2]; };

__device__ __forceinline__ v8bf cvt8_f32_bf16(const float* __restrict__ p) {
  v4f a = *(const v4f*)p;
  v4f b = *(const v4f*)(p + 4);
  v8bf r;
  r[0] = (__bf16)a[0]; r[1] = (__bf16)a[1]; r[2] = (__bf16)a[2]; r[3] = (__bf16)a[3];
  r[4] = (__bf16)b[0]; r[5] = (__bf16)b[1]; r[6] = (__bf16)b[2]; r[7] = (__bf16)b[3];
  return r;
}

__device__ __forceinline__ v8f wmma_bf16(const V16& a, const V16& b, v8f c) {
  return __builtin_amdgcn_wmma_f32_16x16x32_bf16(false, a.v, false, b.v,
                                                 (short)0, c, false, false);
}

// DPP16 row_xmask butterfly (VALU cross-lane, no LDS pipe): reduce over the
// 16-lane row, which is exactly the N extent of a WMMA C tile.
template<int CTRL>
__device__ __forceinline__ float dpp_mov(float x) {
  return __int_as_float(__builtin_amdgcn_update_dpp(
      0, __float_as_int(x), CTRL, 0xf, 0xf, true));
}
__device__ __forceinline__ float rowmax16(float x) {
  x = fmaxf(x, dpp_mov<0x161>(x));   // row_xmask:1
  x = fmaxf(x, dpp_mov<0x162>(x));   // row_xmask:2
  x = fmaxf(x, dpp_mov<0x164>(x));   // row_xmask:4
  x = fmaxf(x, dpp_mov<0x168>(x));   // row_xmask:8
  return x;
}
__device__ __forceinline__ float rowsum16(float x) {
  x += dpp_mov<0x161>(x);
  x += dpp_mov<0x162>(x);
  x += dpp_mov<0x164>(x);
  x += dpp_mov<0x168>(x);
  return x;
}

// Async copy of 16 bytes global -> LDS (ASYNCcnt-tracked, CDNA5 path).
__device__ __forceinline__ void async_load16(unsigned lds_addr, const __bf16* gaddr) {
  asm volatile("global_load_async_to_lds_b128 %0, %1, off"
               :: "v"(lds_addr), "v"(gaddr) : "memory");
}
__device__ __forceinline__ void wait_async0() {
  asm volatile("s_wait_asynccnt 0x0" ::: "memory");
}
__device__ __forceinline__ unsigned lds_addr_of(const void* p) {
  // generic pointer to LDS: addr[31:0] is the LDS offset (ISA 10.2 aperture map)
  return (unsigned)(uintptr_t)p;
}

// ---------------------------------------------------------------------------
// Input projection: out = oscale * act(X @ W^T + bias), X:[R,E] f32, W:[E,E].
// ACT      : logistic map  r_s * s * (1-s),  s = sigmoid(x)
// VT_LAYOUT: write [B,H,D,seq] (for V), else [B,H,seq,D] (for Q,K)
// ---------------------------------------------------------------------------
template<bool ACT, bool VT_LAYOUT>
__launch_bounds__(128)
__global__ void proj_in_kernel(const float* __restrict__ X, const float* __restrict__ W,
                               const float* __restrict__ bias, const float* __restrict__ rp,
                               float oscale, __bf16* __restrict__ out)
{
  const int lane = threadIdx.x & 31;
  const int wave = threadIdx.x >> 5;
  const int lid  = lane & 15;
  const int half = lane >> 4;
  const int row0  = blockIdx.x * 16;
  const int ncol0 = blockIdx.y * 256 + wave * 64;
  const float* xrow = X + (row0 + lid) * E_;

  v8f acc[4] = {};
  for (int kb = 0; kb < E_; kb += 32) {
    const int off = kb + half * 8;              // ISA 16-bit A/B per-lane K pattern
    V16 a;
    a.h[0] = cvt8_f32_bf16(xrow + off);
    a.h[1] = cvt8_f32_bf16(xrow + off + 16);
#pragma unroll
    for (int t = 0; t < 4; ++t) {
      const float* wrow = W + (ncol0 + t * 16 + lid) * E_;   // B[k][n] = W[n][k]
      V16 bm;
      bm.h[0] = cvt8_f32_bf16(wrow + off);
      bm.h[1] = cvt8_f32_bf16(wrow + off + 16);
      acc[t] = wmma_bf16(a, bm, acc[t]);
    }
  }

  const float rsc = ACT ? (4.0f / (1.0f + __expf(-rp[0]))) : 0.0f;
#pragma unroll
  for (int t = 0; t < 4; ++t) {
    const int n  = ncol0 + t * 16 + lid;
    const float bv = bias[n];
    const int h = n >> 6;
    const int d = n & 63;
#pragma unroll
    for (int i = 0; i < 8; ++i) {
      const int r = row0 + i + 8 * half;        // global row in [0, B*T)
      float x = acc[t][i] + bv;
      if (ACT) {
        const float s = 1.0f / (1.0f + __expf(-x));
        x = rsc * s * (1.0f - s);
      }
      x *= oscale;
      const int b  = r >> 11;
      const int tt = r & 2047;
      int idx;
      if (VT_LAYOUT) idx = ((b * H_ + h) * D_ + d) * T_ + tt;   // [B,H,D,S]
      else           idx = ((b * H_ + h) * T_ + tt) * D_ + d;   // [B,H,seq,D]
      out[idx] = (__bf16)x;
    }
  }
}

// ---------------------------------------------------------------------------
// Flash attention. 8 waves/block, all 8 share one (b,h): K/V chunks are
// staged cooperatively into LDS with double-buffered async loads; each wave
// owns a 16-row Q tile. S processed in chunks of 64 columns.
// Q(prescaled):[B,H,T,D], K:[B,H,S,D], Vt:[B,H,D,S] (bf16) -> AT:[B,T,E] bf16
// ---------------------------------------------------------------------------
__launch_bounds__(256)
__global__ void attn_kernel(const __bf16* __restrict__ Q, const __bf16* __restrict__ K,
                            const __bf16* __restrict__ Vt, __bf16* __restrict__ AT)
{
  __shared__ __align__(16) __bf16 Ks[2][64][64];     // [buf][s_local][d]
  __shared__ __align__(16) __bf16 Vs[2][64][64];     // [buf][d][s_local]
  __shared__ __align__(16) __bf16 plds[8][1024];     // per-wave P in A-reg layout

  const int tid  = threadIdx.x;
  const int lane = tid & 31;
  const int wave = tid >> 5;
  const int lid  = lane & 15;
  const int half = lane >> 4;
  const int aoff = half * 8;
  const int bh = blockIdx.x >> 4;                    // all 8 waves share (b,h)
  const int qt = (blockIdx.x & 15) * 8 + wave;
  const __bf16* Qb = Q  + (bh * T_ + qt * 16) * D_;
  const __bf16* Kb = K  + bh * T_ * D_;
  const __bf16* Vb = Vt + bh * D_ * T_;

  // Q tile -> A registers (two K=32 chunks over D=64); Q already holds 1/sqrt(D)
  V16 qa0, qa1;
  {
    const __bf16* qrow = Qb + lid * D_;
    qa0.h[0] = *(const v8bf*)(qrow + aoff);
    qa0.h[1] = *(const v8bf*)(qrow + aoff + 16);
    qa1.h[0] = *(const v8bf*)(qrow + 32 + aoff);
    qa1.h[1] = *(const v8bf*)(qrow + 32 + aoff + 16);
  }

  v8f o[4] = {};
  float mrow[8], lrow[8];
#pragma unroll
  for (int i = 0; i < 8; ++i) { mrow[i] = -3.0e38f; lrow[i] = 0.0f; }
  __bf16* pw = &plds[wave][0];

  // cooperative async stage of one 64-column chunk (8 KB K + 8 KB V):
  // 512 x 16B transfers -> 2 K-lines + 2 V-lines per thread.
  auto stage = [&](int buf, int s0) {
#pragma unroll
    for (int u = 0; u < 2; ++u) {
      const int idx = tid * 2 + u;
      const int r   = idx >> 3;          // 0..63
      const int seg = (idx & 7) * 8;     // bf16 offset within 128B row
      async_load16(lds_addr_of(&Ks[buf][r][seg]), Kb + (s0 + r) * D_ + seg);
      async_load16(lds_addr_of(&Vs[buf][r][seg]), Vb + r * T_ + s0 + seg);
    }
  };

  stage(0, 0);
  const int NIT = T_ / 64;                           // 32
  for (int it = 0; it < NIT; ++it) {
    const int buf = it & 1;
    wait_async0();                                   // my stage(buf) complete
    __syncthreads();                                 // everyone's stage(buf) done,
                                                     // everyone done reading buf^1
    if (it + 1 < NIT) stage(buf ^ 1, (it + 1) * 64); // overlap next DMA with math

    // ---- scores: four 16x16 tiles over 64 new columns ---------------------
    v8f c[4] = {};
#pragma unroll
    for (int tt = 0; tt < 4; ++tt) {
      const __bf16* krow = &Ks[buf][tt * 16 + lid][0];
      V16 b0, b1;
      b0.h[0] = *(const v8bf*)(krow + aoff);
      b0.h[1] = *(const v8bf*)(krow + aoff + 16);
      b1.h[0] = *(const v8bf*)(krow + 32 + aoff);
      b1.h[1] = *(const v8bf*)(krow + 32 + aoff + 16);
      c[tt] = wmma_bf16(qa0, b0, c[tt]);
      c[tt] = wmma_bf16(qa1, b1, c[tt]);
    }

    // ---- online softmax over 64 columns (DPP row reductions) --------------
#pragma unroll
    for (int i = 0; i < 8; ++i) {
      const float x0 = c[0][i], x1 = c[1][i], x2 = c[2][i], x3 = c[3][i];
      float mx = rowmax16(fmaxf(fmaxf(x0, x1), fmaxf(x2, x3)));
      const float mn   = fmaxf(mrow[i], mx);
      const float corr = __expf(mrow[i] - mn);
      mrow[i] = mn;
      const float p0 = __expf(x0 - mn);
      const float p1 = __expf(x1 - mn);
      const float p2 = __expf(x2 - mn);
      const float p3 = __expf(x3 - mn);
      const float rs = rowsum16((p0 + p1) + (p2 + p3));
      lrow[i] = lrow[i] * corr + rs;
      o[0][i] *= corr; o[1][i] *= corr; o[2][i] *= corr; o[3][i] *= corr;

      // scatter P (C layout) -> A-matrix register layout in LDS.
      // chunk c covers s_local = 32c..32c+31; within chunk:
      //   lane = r + 16*bit3(s32), slot = (s32&7) + 8*(s32>>4)
      const int tl = (i + 8 * half) + 16 * ((lid >> 3) & 1);
      const int sl = lid & 7;
      __bf16* pd = pw + tl * 32;
      pd[sl]          = (__bf16)p0;   // chunk0, tile0
      pd[sl + 8]      = (__bf16)p1;   // chunk0, tile1
      pd[16 + sl]     = (__bf16)p2;   // chunk1, tile2
      pd[16 + sl + 8] = (__bf16)p3;   // chunk1, tile3
    }

    // ---- O += P(16x64) * V(64x64) -----------------------------------------
    V16 pa0, pa1;
    pa0.h[0] = *(const v8bf*)(pw + lane * 32);
    pa0.h[1] = *(const v8bf*)(pw + lane * 32 + 8);
    pa1.h[0] = *(const v8bf*)(pw + lane * 32 + 16);
    pa1.h[1] = *(const v8bf*)(pw + lane * 32 + 24);
#pragma unroll
    for (int j = 0; j < 4; ++j) {
      const __bf16* vrow = &Vs[buf][j * 16 + lid][0];
      V16 vb0, vb1;
      vb0.h[0] = *(const v8bf*)(vrow + aoff);
      vb0.h[1] = *(const v8bf*)(vrow + aoff + 16);
      vb1.h[0] = *(const v8bf*)(vrow + 32 + aoff);
      vb1.h[1] = *(const v8bf*)(vrow + 32 + aoff + 16);
      o[j] = wmma_bf16(pa0, vb0, o[j]);
      o[j] = wmma_bf16(pa1, vb1, o[j]);
    }
    __syncthreads();                                 // done reading buf
  }

  // ---- normalize and write [B,T,E] (head-concat) --------------------------
  const int b = bh >> 3;
  const int h = bh & 7;
#pragma unroll
  for (int i = 0; i < 8; ++i) {
    const float rl = 1.0f / lrow[i];
    const int r = qt * 16 + i + 8 * half;
    const int base = (b * T_ + r) * E_ + h * D_;
#pragma unroll
    for (int j = 0; j < 4; ++j)
      AT[base + j * 16 + lid] = (__bf16)(o[j][i] * rl);
  }
}

// ---------------------------------------------------------------------------
// Output projection: out = X @ Wo^T + bo, X:[R,E] bf16, out f32
// ---------------------------------------------------------------------------
__launch_bounds__(128)
__global__ void proj_out_kernel(const __bf16* __restrict__ X, const float* __restrict__ W,
                                const float* __restrict__ bias, float* __restrict__ out)
{
  const int lane = threadIdx.x & 31;
  const int wave = threadIdx.x >> 5;
  const int lid  = lane & 15;
  const int half = lane >> 4;
  const int row0  = blockIdx.x * 16;
  const int ncol0 = blockIdx.y * 256 + wave * 64;
  const __bf16* xrow = X + (row0 + lid) * E_;

  v8f acc[4] = {};
  for (int kb = 0; kb < E_; kb += 32) {
    const int off = kb + half * 8;
    V16 a;
    a.h[0] = *(const v8bf*)(xrow + off);
    a.h[1] = *(const v8bf*)(xrow + off + 16);
#pragma unroll
    for (int t = 0; t < 4; ++t) {
      const float* wrow = W + (ncol0 + t * 16 + lid) * E_;
      V16 bm;
      bm.h[0] = cvt8_f32_bf16(wrow + off);
      bm.h[1] = cvt8_f32_bf16(wrow + off + 16);
      acc[t] = wmma_bf16(a, bm, acc[t]);
    }
  }

#pragma unroll
  for (int t = 0; t < 4; ++t) {
    const int n  = ncol0 + t * 16 + lid;
    const float bv = bias[n];
#pragma unroll
    for (int i = 0; i < 8; ++i) {
      const int r = row0 + i + 8 * half;
      out[r * E_ + n] = acc[t][i] + bv;
    }
  }
}

// ---------------------------------------------------------------------------
extern "C" void kernel_launch(void* const* d_in, const int* in_sizes, int n_in,
                              void* d_out, int out_size, void* d_ws, size_t ws_size,
                              hipStream_t stream) {
  (void)in_sizes; (void)n_in; (void)out_size; (void)ws_size;
  const float* query = (const float*)d_in[0];
  const float* key_  = (const float*)d_in[1];
  const float* value = (const float*)d_in[2];
  const float* Wq = (const float*)d_in[3];
  const float* bq = (const float*)d_in[4];
  const float* Wk = (const float*)d_in[5];
  const float* bk = (const float*)d_in[6];
  const float* Wv = (const float*)d_in[7];
  const float* bv = (const float*)d_in[8];
  const float* Wo = (const float*)d_in[9];
  const float* bo = (const float*)d_in[10];
  const float* r  = (const float*)d_in[11];

  const size_t NTOK = (size_t)4 * 2048 * 512;   // elements per bf16 tensor
  __bf16* Qb  = (__bf16*)d_ws;
  __bf16* Kb  = Qb  + NTOK;
  __bf16* Vtb = Kb  + NTOK;
  __bf16* AT  = Vtb + NTOK;                     // 32 MB bf16 workspace total

  dim3 pgrid(512, 2);   // (B*T/16, E/256)
  dim3 pblock(128);
  // fold 1/sqrt(D) into Q at projection time
  proj_in_kernel<true,  false><<<pgrid, pblock, 0, stream>>>(query, Wq, bq, r, 0.125f, Qb);
  proj_in_kernel<true,  false><<<pgrid, pblock, 0, stream>>>(key_,  Wk, bk, r, 1.0f,   Kb);
  proj_in_kernel<false, true ><<<pgrid, pblock, 0, stream>>>(value, Wv, bv, r, 1.0f,   Vtb);

  // 32 (b,h) x 128 q-tiles; 8 q-tiles (one per wave) per block
  attn_kernel<<<512, 256, 0, stream>>>(Qb, Kb, Vtb, AT);

  proj_out_kernel<<<pgrid, pblock, 0, stream>>>(AT, Wo, bo, (float*)d_out);
}
